// MultiHeadAttention_13159779795587
// MI455X (gfx1250) — matrix operand order repaired and checked by reading, not compiled, not run on hardware
//
#include <hip/hip_runtime.h>
#include <hip/hip_bf16.h>
#include <stdint.h>

// ---------------------------------------------------------------------------
// MI455X (gfx1250, wave32) multi-head attention, all matmuls on
// v_wmma_f32_16x16x32_bf16, GEMM staging via async global->LDS copies
// (double buffered, ASYNCcnt-tracked).
// ---------------------------------------------------------------------------

typedef __attribute__((ext_vector_type(16))) __bf16 v16bf;
typedef __attribute__((ext_vector_type(8)))  float  v8f;

// pointer types matching the async builtin's prototype:
// (v4i32 AS1* gsrc, v4i32 AS3* ldsdst, imm offset, imm cpol)
typedef int v4i32 __attribute__((vector_size(16)));
typedef v4i32 __attribute__((address_space(1)))* g_v4i_ptr;
typedef v4i32 __attribute__((address_space(3)))* l_v4i_ptr;

#define D_MODEL   1024
#define NUM_HEADS 16
#define D_K       64

// ---------------- async global->LDS path detection -------------------------
#if defined(__HIP_DEVICE_COMPILE__) && __has_builtin(__builtin_amdgcn_global_load_async_to_lds_b128)
#define USE_ASYNC_LDS 1
#pragma message("CDNA5: using __builtin_amdgcn_global_load_async_to_lds_b128")
#else
#define USE_ASYNC_LDS 0
#if defined(__HIP_DEVICE_COMPILE__)
#pragma message("CDNA5: async-to-LDS builtin NOT available, using sync staging")
#endif
#endif

#if USE_ASYNC_LDS
#if __has_builtin(__builtin_amdgcn_s_wait_asynccnt)
#define WAIT_ASYNC(n) __builtin_amdgcn_s_wait_asynccnt(n)
#pragma message("CDNA5: using __builtin_amdgcn_s_wait_asynccnt")
#else
#define WAIT_ASYNC(n) asm volatile("s_wait_asynccnt %0" ::"n"(n) : "memory")
#pragma message("CDNA5: using inline-asm s_wait_asynccnt")
#endif
#else
#define WAIT_ASYNC(n) ((void)0)
#endif

// copy 16 bytes global -> LDS (async engine if available)
__device__ __forceinline__ void cp16_g2l(const void* g, void* l) {
#if USE_ASYNC_LDS
  __builtin_amdgcn_global_load_async_to_lds_b128(
      (g_v4i_ptr)(uintptr_t)g,
      (l_v4i_ptr)(uint32_t)(uintptr_t)l,
      0, 0);
#else
  *(uint4*)l = *(const uint4*)g;
#endif
}

__device__ __forceinline__ v8f wmma_bf16(v16bf a, v16bf b, v8f c) {
  return __builtin_amdgcn_wmma_f32_16x16x32_bf16(false, a, false, b,
                                                 (short)0, c, false, false);
}

// A-fragment (16x32 bf16) K index for element i, per ISA VGPR layout.
__device__ __forceinline__ int kA(int i, int lanehi) {
  return (i < 8) ? (lanehi * 8 + i) : (16 + lanehi * 8 + (i - 8));
}

// ---------------------------------------------------------------------------
// fp32 -> bf16 conversion pre-pass (one shot; makes GEMM inputs byte-copyable
// by the async-to-LDS engine and halves projection-GEMM HBM traffic).
// ---------------------------------------------------------------------------
__global__ __launch_bounds__(256)
void f32_to_bf16(const float* __restrict__ src, __bf16* __restrict__ dst, int n) {
  int i = (blockIdx.x * 256 + threadIdx.x) * 4;
  if (i + 3 < n) {
    float4 v = *(const float4*)(src + i);
    dst[i + 0] = (__bf16)v.x;
    dst[i + 1] = (__bf16)v.y;
    dst[i + 2] = (__bf16)v.z;
    dst[i + 3] = (__bf16)v.w;
  }
}

// ---------------------------------------------------------------------------
// GEMM: C[M,N] = A[M,K](bf16) * W[N,K](bf16)^T + bias[N](f32)
// 256 threads = 8 waves (4x2); block tile 128x128, BK=64, double-buffered LDS.
// Each wave: 32x64 tile = 2x4 WMMA accumulators; 16 WMMAs per K-slab.
// ---------------------------------------------------------------------------
template <typename OT>
__global__ __launch_bounds__(256)
void gemm_bf16_xwt(const __bf16* __restrict__ A, const __bf16* __restrict__ W,
                   const float* __restrict__ bias, OT* __restrict__ C,
                   int M, int N, int K) {
  __shared__ __attribute__((aligned(16))) __bf16 As[2][128][64];
  __shared__ __attribute__((aligned(16))) __bf16 Bs[2][128][64];

  const int tid    = threadIdx.x;
  const int lane   = tid & 31;
  const int wave   = tid >> 5;
  const int lanelo = lane & 15;
  const int lanehi = lane >> 4;
  const int wm     = wave >> 1;           // 0..3
  const int wn     = wave & 1;            // 0..1
  const int m0     = blockIdx.y * 128;
  const int n0     = blockIdx.x * 128;

  v8f acc[2][4] = {};

  // stage a 128x64 bf16 slab of A and W into LDS buffer `b`
  auto stage = [&](int b, int k0) {
#pragma unroll
    for (int j = 0; j < 4; ++j) {
      int c  = tid + 256 * j;             // 0..1023 16B-chunks
      int r  = c >> 3;                    // 0..127
      int ch = (c & 7) * 8;               // bf16 col 0,8,...,56
      cp16_g2l(A + (size_t)(m0 + r) * K + k0 + ch, &As[b][r][ch]);
      cp16_g2l(W + (size_t)(n0 + r) * K + k0 + ch, &Bs[b][r][ch]);
    }
  };

  stage(0, 0);
  int buf = 0;
  const int nk = K / 64;
  for (int kt = 0; kt < nk; ++kt) {
    if (kt + 1 < nk) {
      stage(buf ^ 1, (kt + 1) * 64);      // issue next slab (async)
      WAIT_ASYNC(8);                      // wait for current slab's 8 issues
    } else {
      WAIT_ASYNC(0);
    }
    __syncthreads();

#pragma unroll
    for (int kk = 0; kk < 2; ++kk) {      // two 32-wide WMMA K-steps
      v16bf af[2], bfr[4];
#pragma unroll
      for (int mi = 0; mi < 2; ++mi)
#pragma unroll
        for (int i = 0; i < 16; ++i)
          af[mi][i] = As[buf][wm * 32 + mi * 16 + lanelo][kk * 32 + kA(i, lanehi)];
#pragma unroll
      for (int ni = 0; ni < 4; ++ni)
#pragma unroll
        for (int i = 0; i < 16; ++i)
          bfr[ni][i] = Bs[buf][wn * 64 + ni * 16 + lanelo][kk * 32 + kA(i, lanehi)];
#pragma unroll
      for (int mi = 0; mi < 2; ++mi)
#pragma unroll
        for (int ni = 0; ni < 4; ++ni)
          acc[mi][ni] = wmma_bf16(af[mi], bfr[ni], acc[mi][ni]);
    }
    __syncthreads();
    buf ^= 1;
  }

  // epilogue: bias + store (C/D layout: m = r + 8*lanehi, n = lanelo)
#pragma unroll
  for (int mi = 0; mi < 2; ++mi)
#pragma unroll
    for (int ni = 0; ni < 4; ++ni) {
      int gn   = n0 + wn * 64 + ni * 16 + lanelo;
      float bb = bias[gn];
#pragma unroll
      for (int r = 0; r < 8; ++r) {
        int gm = m0 + wm * 32 + mi * 16 + r + 8 * lanehi;
        C[(size_t)gm * N + gn] = (OT)(acc[mi][ni][r] + bb);
      }
    }
}

// ---------------------------------------------------------------------------
// Flash attention: one wave per (b, h, 16-query block). Q,K,V bf16.
// ---------------------------------------------------------------------------
__global__ __launch_bounds__(32)
void attn_kernel(const __bf16* __restrict__ Q, const __bf16* __restrict__ Km,
                 const __bf16* __restrict__ Vm, __bf16* __restrict__ ctx,
                 int S) {
  const int lane   = threadIdx.x;
  const int lanelo = lane & 15;
  const int lanehi = lane >> 4;

  int idx        = blockIdx.x;
  const int nqb  = S / 16;
  const int qblk = idx % nqb;        idx /= nqb;
  const int h    = idx % NUM_HEADS;  idx /= NUM_HEADS;
  const int b    = idx;
  const int q0   = qblk * 16;

  const size_t base = ((size_t)b * S) * D_MODEL + (size_t)h * D_K;
  const __bf16* Qb = Q  + base;
  const __bf16* Kb = Km + base;
  const __bf16* Vb = Vm + base;

  __shared__ __bf16 Ps[16][32];

  // Q fragments, pre-scaled by 1/sqrt(Dk)
  v16bf qf[2];
#pragma unroll
  for (int kf = 0; kf < 2; ++kf)
#pragma unroll
    for (int i = 0; i < 16; ++i) {
      int d = kf * 32 + kA(i, lanehi);
      qf[kf][i] = (__bf16)((float)Qb[(size_t)(q0 + lanelo) * D_MODEL + d] * 0.125f);
    }

  float mrun[8], lrun[8];
  v8f   o[4] = {};
#pragma unroll
  for (int r = 0; r < 8; ++r) { mrun[r] = -1e30f; lrun[r] = 0.0f; }

  for (int kb = 0; kb < S; kb += 32) {
    if (kb + 32 < S) {
      __builtin_prefetch(&Kb[(size_t)(kb + 32) * D_MODEL], 0, 1);
      __builtin_prefetch(&Vb[(size_t)(kb + 32) * D_MODEL], 0, 1);
    }

    // scores: 16x32 tile = 2 n-tiles x 2 K-fragments
    v8f sc[2] = {};
#pragma unroll
    for (int nt = 0; nt < 2; ++nt) {
#pragma unroll
      for (int kf = 0; kf < 2; ++kf) {
        v16bf kfr;
#pragma unroll
        for (int i = 0; i < 16; ++i) {
          int d = kf * 32 + kA(i, lanehi);
          kfr[i] = Kb[(size_t)(kb + nt * 16 + lanelo) * D_MODEL + d];
        }
        sc[nt] = wmma_bf16(qf[kf], kfr, sc[nt]);
      }
    }

    // streaming softmax per row (m = r + 8*lanehi)
    float pr0[8], pr1[8], alpha[8];
#pragma unroll
    for (int r = 0; r < 8; ++r) {
      float mx = fmaxf(sc[0][r], sc[1][r]);
      mx = fmaxf(mx, __shfl_xor(mx, 1));
      mx = fmaxf(mx, __shfl_xor(mx, 2));
      mx = fmaxf(mx, __shfl_xor(mx, 4));
      mx = fmaxf(mx, __shfl_xor(mx, 8));
      float mnew = fmaxf(mrun[r], mx);
      float a    = __expf(mrun[r] - mnew);
      float p0   = __expf(sc[0][r] - mnew);
      float p1   = __expf(sc[1][r] - mnew);
      float rs   = p0 + p1;
      rs += __shfl_xor(rs, 1);
      rs += __shfl_xor(rs, 2);
      rs += __shfl_xor(rs, 4);
      rs += __shfl_xor(rs, 8);
      lrun[r]  = lrun[r] * a + rs;
      mrun[r]  = mnew;
      alpha[r] = a;
      pr0[r]   = p0;
      pr1[r]   = p1;
    }

    // re-layout P (D-layout) -> A-fragment layout via LDS
#pragma unroll
    for (int r = 0; r < 8; ++r) {
      Ps[r + 8 * lanehi][lanelo]      = (__bf16)pr0[r];
      Ps[r + 8 * lanehi][16 + lanelo] = (__bf16)pr1[r];
    }
    __syncthreads();
    v16bf pf;
#pragma unroll
    for (int i = 0; i < 16; ++i)
      pf[i] = Ps[lanelo][kA(i, lanehi)];

    // rescale running context, then P @ V
#pragma unroll
    for (int t = 0; t < 4; ++t)
#pragma unroll
      for (int r = 0; r < 8; ++r)
        o[t][r] *= alpha[r];

#pragma unroll
    for (int t = 0; t < 4; ++t) {
      v16bf vf;
#pragma unroll
      for (int i = 0; i < 16; ++i)
        vf[i] = Vb[(size_t)(kb + kA(i, lanehi)) * D_MODEL + t * 16 + lanelo];
      o[t] = wmma_bf16(pf, vf, o[t]);
    }
    __syncthreads();
  }

  // normalize + store ctx (bf16, [B,S,H*Dk])
  __bf16* Cb = ctx + base + (size_t)q0 * D_MODEL;
#pragma unroll
  for (int t = 0; t < 4; ++t)
#pragma unroll
    for (int r = 0; r < 8; ++r) {
      int m = r + 8 * lanehi;
      Cb[(size_t)m * D_MODEL + t * 16 + lanelo] = (__bf16)(o[t][r] / lrun[r]);
    }
}

// ---------------------------------------------------------------------------
extern "C" void kernel_launch(void* const* d_in, const int* in_sizes, int n_in,
                              void* d_out, int out_size, void* d_ws, size_t ws_size,
                              hipStream_t stream) {
  const float* query = (const float*)d_in[0];
  const float* key   = (const float*)d_in[1];
  const float* value = (const float*)d_in[2];
  const float* Wq    = (const float*)d_in[3];
  const float* bq    = (const float*)d_in[4];
  const float* Wk    = (const float*)d_in[5];
  const float* bk    = (const float*)d_in[6];
  const float* Wv    = (const float*)d_in[7];
  const float* bv    = (const float*)d_in[8];
  const float* Wo    = (const float*)d_in[9];
  const float* bo    = (const float*)d_in[10];

  const int Bsz = 4, S = 2048;
  const int M  = Bsz * S;                  // 8192
  const int D  = D_MODEL;                  // 1024
  const size_t MD = (size_t)M * D;         // 8.39M elems
  const size_t DD = (size_t)D * D;         // 1.05M elems

  // ws layout (bf16): inputs q/k/v, weights wq/wk/wv/wo, Qp/Kp/Vp/ctx
  __bf16* qb  = (__bf16*)d_ws;
  __bf16* kb  = qb  + MD;
  __bf16* vb  = kb  + MD;
  __bf16* wqb = vb  + MD;
  __bf16* wkb = wqb + DD;
  __bf16* wvb = wkb + DD;
  __bf16* wob = wvb + DD;
  __bf16* Qp  = wob + DD;
  __bf16* Kp  = Qp  + MD;
  __bf16* Vp  = Kp  + MD;
  __bf16* Cp  = Vp  + MD;

  // 1) fp32 -> bf16 conversion pre-pass
  {
    int na = (int)MD, nw = (int)DD;
    dim3 ba((na / 4 + 255) / 256), bw((nw / 4 + 255) / 256), tb(256);
    f32_to_bf16<<<ba, tb, 0, stream>>>(query, qb, na);
    f32_to_bf16<<<ba, tb, 0, stream>>>(key,   kb, na);
    f32_to_bf16<<<ba, tb, 0, stream>>>(value, vb, na);
    f32_to_bf16<<<bw, tb, 0, stream>>>(Wq, wqb, nw);
    f32_to_bf16<<<bw, tb, 0, stream>>>(Wk, wkb, nw);
    f32_to_bf16<<<bw, tb, 0, stream>>>(Wv, wvb, nw);
    f32_to_bf16<<<bw, tb, 0, stream>>>(Wo, wob, nw);
  }

  // 2) Q/K/V projections (bf16 out)
  dim3 gg(D / 128, M / 128);   // (8, 64)
  dim3 gb(256);
  gemm_bf16_xwt<__bf16><<<gg, gb, 0, stream>>>(qb, wqb, bq, Qp, M, D, D);
  gemm_bf16_xwt<__bf16><<<gg, gb, 0, stream>>>(kb, wkb, bk, Kp, M, D, D);
  gemm_bf16_xwt<__bf16><<<gg, gb, 0, stream>>>(vb, wvb, bv, Vp, M, D, D);

  // 3) flash attention
  attn_kernel<<<dim3(Bsz * NUM_HEADS * (S / 16)), dim3(32), 0, stream>>>(Qp, Kp, Vp, Cp, S);

  // 4) output projection (fp32 out)
  gemm_bf16_xwt<float><<<gg, gb, 0, stream>>>(Cp, wob, bo, (float*)d_out, M, D, D);
}
